// TopologyPrompt_34248069218352
// MI455X (gfx1250) — compile-verified, matrix operand order checked by reading, not run on hardware
//
#include <hip/hip_runtime.h>
#include <hip/hip_bf16.h>
#include <math.h>

// ---------------- CDNA5 WMMA types ----------------
typedef __attribute__((ext_vector_type(16))) __bf16 v16bf;
typedef __attribute__((ext_vector_type(8)))  float  v8f;

#define DIM       128
#define DIM2      256
#define N_ANCH    5
#define HID       64
#define NEG_SLOPE 0.01f

// hash-based uniform in [0,1) (stand-in for jax threefry; no runtime check here)
__device__ __forceinline__ float hash_uniform(unsigned i, unsigned seed) {
  unsigned h = i * 0x9E3779B9u ^ seed;
  h ^= h >> 16; h *= 0x85EBCA6Bu;
  h ^= h >> 13; h *= 0xC2B2AE35u;
  h ^= h >> 16;
  return (float)(h >> 8) * (1.0f / 16777216.0f);
}

// ---------------- zero workspace ----------------
__global__ void zero_kernel(float* __restrict__ p, size_t n) {
  size_t i = (size_t)blockIdx.x * blockDim.x + threadIdx.x;
  size_t stride = (size_t)gridDim.x * blockDim.x;
  for (; i < n; i += stride) p[i] = 0.0f;
}

// ---------------- degree = segment_sum(ones, src) ----------------
__global__ void degree_kernel(const int* __restrict__ src, float* __restrict__ degree, int E) {
  int i = blockIdx.x * blockDim.x + threadIdx.x;
  int stride = gridDim.x * blockDim.x;
  for (; i < E; i += stride) atomicAdd(&degree[src[i]], 1.0f);
}

// ---------------- degree max / sum ----------------
__global__ void stats_kernel(const float* __restrict__ degree, float* __restrict__ stats, int N) {
  __shared__ float smax[256];
  __shared__ float ssum[256];
  int t = threadIdx.x;
  float mx = 0.0f, sm = 0.0f;
  for (int i = blockIdx.x * blockDim.x + t; i < N; i += gridDim.x * blockDim.x) {
    float d = degree[i];
    mx = fmaxf(mx, d);
    sm += d;
  }
  smax[t] = mx; ssum[t] = sm;
  __syncthreads();
  for (int o = 128; o > 0; o >>= 1) {
    if (t < o) { smax[t] = fmaxf(smax[t], smax[t + o]); ssum[t] += ssum[t + o]; }
    __syncthreads();
  }
  if (t == 0) {
    // degree >= 0, so uint bit-pattern max == float max
    atomicMax((unsigned int*)&stats[0], __float_as_uint(smax[0]));
    atomicAdd(&stats[1], ssum[0]);
  }
}

// ---------------- NodePromptplus: softmax(x@Wa^T+b) @ node_anchor + x ----------------
// block = 128 threads (4 waves); each wave owns a 16-node WMMA tile; block tile = 64 nodes
__global__ void node_kernel(const float* __restrict__ x,
                            const float* __restrict__ attn_weight,  // [5,128]
                            const float* __restrict__ attn_bias,    // [5]
                            const float* __restrict__ node_anchor,  // [5,128]
                            float* __restrict__ node_prompted,      // [N,128]
                            int N) {
  __shared__ float lds_logits[64][16];
  __shared__ float lds_b[64][8];
  const int t    = threadIdx.x;
  const int lane = t & 31;
  const int wave = t >> 5;
  const int M    = lane & 15;
  const int half = lane >> 4;
  const int ncol = lane & 15;

  // B fragments: attn_weight^T padded to 128x16 (cols >=5 are zero). Hoisted.
  v16bf Bf[4];
#pragma unroll
  for (int kb = 0; kb < 4; ++kb) {
    v16bf b;
#pragma unroll
    for (int e = 0; e < 16; ++e) {
      int Kg = kb * 32 + (e & 7) + ((e >> 3) << 4) + half * 8;
      float v = (ncol < N_ANCH) ? attn_weight[ncol * DIM + Kg] : 0.0f;
      b[e] = (__bf16)v;
    }
    Bf[kb] = b;
  }
  float na[N_ANCH], bias[N_ANCH];
#pragma unroll
  for (int a = 0; a < N_ANCH; ++a) {
    na[a]   = node_anchor[a * DIM + t];
    bias[a] = attn_bias[a];
  }

  const int ntiles = (N + 63) / 64;
  for (int tile = blockIdx.x; tile < ntiles; tile += gridDim.x) {
    int row = tile * 64 + wave * 16 + M;
    if (row >= N) row = N - 1;                 // clamp: keep EXEC full for WMMA
    const float* xr = x + (size_t)row * DIM;

    v8f c = {};
#pragma unroll
    for (int kb = 0; kb < 4; ++kb) {
      v16bf a;
#pragma unroll
      for (int r = 0; r < 2; ++r) {
        int Kg0 = kb * 32 + r * 16 + half * 8;
        float4 u = *(const float4*)(xr + Kg0);
        float4 v = *(const float4*)(xr + Kg0 + 4);
        a[r*8+0]=(__bf16)u.x; a[r*8+1]=(__bf16)u.y; a[r*8+2]=(__bf16)u.z; a[r*8+3]=(__bf16)u.w;
        a[r*8+4]=(__bf16)v.x; a[r*8+5]=(__bf16)v.y; a[r*8+6]=(__bf16)v.z; a[r*8+7]=(__bf16)v.w;
      }
      c = __builtin_amdgcn_wmma_f32_16x16x32_bf16(false, a, false, Bf[kb], (short)0, c, false, false);
    }
#pragma unroll
    for (int r = 0; r < 8; ++r)
      lds_logits[wave * 16 + r + 8 * half][lane & 15] = c[r];
    __syncthreads();

    if (t < 64) {
      float l[N_ANCH];
      float mx = -1e30f;
#pragma unroll
      for (int a = 0; a < N_ANCH; ++a) { l[a] = lds_logits[t][a] + bias[a]; mx = fmaxf(mx, l[a]); }
      float sum = 0.0f;
#pragma unroll
      for (int a = 0; a < N_ANCH; ++a) { l[a] = __expf(l[a] - mx); sum += l[a]; }
      float inv = 1.0f / sum;
#pragma unroll
      for (int a = 0; a < N_ANCH; ++a) lds_b[t][a] = l[a] * inv;
    }
    __syncthreads();

    // column-parallel: thread t = column t, coalesced across the 64 nodes
    int base = tile * 64;
    for (int m = 0; m < 64; ++m) {
      int n = base + m;
      if (n >= N) break;
      float acc = 0.0f;
#pragma unroll
      for (int a = 0; a < N_ANCH; ++a) acc = fmaf(lds_b[m][a], na[a], acc);
      node_prompted[(size_t)n * DIM + t] = x[(size_t)n * DIM + t] + acc;
    }
    __syncthreads();
  }
}

// ---------------- EdgePromptplus + scatter ----------------
// block = 128 threads (4 waves); block tile = 64 edges, wave tile = 16 edges, K = 256
__global__ void edge_kernel(const float* __restrict__ x,
                            const int* __restrict__ edge_src,
                            const int* __restrict__ edge_dst,
                            const float* __restrict__ w_weight,      // [5,256]
                            const float* __restrict__ w_bias,        // [5]
                            const float* __restrict__ anchor_prompt, // [5,128]
                            float* __restrict__ edge_prompt,         // [E,128] (output, NT stores)
                            float* __restrict__ edge_agg,            // [N,128] (atomic scatter)
                            int E) {
  __shared__ float lds_logits[64][16];
  __shared__ float lds_b[64][8];
  __shared__ int   lds_src[64];
  const int t    = threadIdx.x;
  const int lane = t & 31;
  const int wave = t >> 5;
  const int M    = lane & 15;
  const int half = lane >> 4;
  const int ncol = lane & 15;

  // B fragments: w_weight^T padded to 256x16. Hoisted out of the tile loop.
  v16bf Bf[8];
#pragma unroll
  for (int kb = 0; kb < 8; ++kb) {
    v16bf b;
#pragma unroll
    for (int e = 0; e < 16; ++e) {
      int Kg = kb * 32 + (e & 7) + ((e >> 3) << 4) + half * 8;
      float v = (ncol < N_ANCH) ? w_weight[ncol * DIM2 + Kg] : 0.0f;
      b[e] = (__bf16)v;
    }
    Bf[kb] = b;
  }
  float ap[N_ANCH], bias[N_ANCH];
#pragma unroll
  for (int a = 0; a < N_ANCH; ++a) {
    ap[a]   = anchor_prompt[a * DIM + t];
    bias[a] = w_bias[a];
  }

  const int ntiles = (E + 63) / 64;
  for (int tile = blockIdx.x; tile < ntiles; tile += gridDim.x) {
    int eM = tile * 64 + wave * 16 + M;
    if (eM >= E) eM = E - 1;                   // clamp: EXEC stays full for WMMA
    int s = edge_src[eM];
    int d = edge_dst[eM];
    if (half == 0) lds_src[wave * 16 + M] = s;
    const float* xs = x + (size_t)s * DIM;
    const float* xd = x + (size_t)d * DIM;

    v8f c = {};
#pragma unroll
    for (int kb = 0; kb < 8; ++kb) {
      v16bf a;
#pragma unroll
      for (int r = 0; r < 2; ++r) {
        int Kg0 = kb * 32 + r * 16 + half * 8;        // multiple of 8: run never crosses 128
        const float* p = (Kg0 < DIM) ? (xs + Kg0) : (xd + (Kg0 - DIM));
        float4 u = *(const float4*)p;
        float4 v = *(const float4*)(p + 4);
        a[r*8+0]=(__bf16)u.x; a[r*8+1]=(__bf16)u.y; a[r*8+2]=(__bf16)u.z; a[r*8+3]=(__bf16)u.w;
        a[r*8+4]=(__bf16)v.x; a[r*8+5]=(__bf16)v.y; a[r*8+6]=(__bf16)v.z; a[r*8+7]=(__bf16)v.w;
      }
      c = __builtin_amdgcn_wmma_f32_16x16x32_bf16(false, a, false, Bf[kb], (short)0, c, false, false);
    }
#pragma unroll
    for (int r = 0; r < 8; ++r)
      lds_logits[wave * 16 + r + 8 * half][lane & 15] = c[r];
    __syncthreads();

    if (t < 64) {
      float l[N_ANCH];
      float mx = -1e30f;
#pragma unroll
      for (int a = 0; a < N_ANCH; ++a) {
        float v = lds_logits[t][a] + bias[a];
        v = (v > 0.0f) ? v : NEG_SLOPE * v;     // leaky_relu
        l[a] = v;
        mx = fmaxf(mx, v);
      }
      float sum = 0.0f;
#pragma unroll
      for (int a = 0; a < N_ANCH; ++a) { l[a] = __expf(l[a] - mx); sum += l[a]; }
      float inv = 1.0f / sum;
#pragma unroll
      for (int a = 0; a < N_ANCH; ++a) lds_b[t][a] = l[a] * inv;
    }
    __syncthreads();

    // column-parallel b @ anchor_prompt: NT store (409MB stream, keep L2 for x),
    // plus f32 atomic scatter into L2-resident edge_agg
    int base = tile * 64;
    for (int m = 0; m < 64; ++m) {
      int e = base + m;
      if (e >= E) break;
      float acc = 0.0f;
#pragma unroll
      for (int a = 0; a < N_ANCH; ++a) acc = fmaf(lds_b[m][a], ap[a], acc);
      __builtin_nontemporal_store(acc, &edge_prompt[(size_t)e * DIM + t]);
      atomicAdd(&edge_agg[(size_t)lds_src[m] * DIM + t], acc);
    }
    __syncthreads();
  }
}

// ---------------- topology MLP + final combine ----------------
__global__ void topo_final_kernel(const float* __restrict__ degree,
                                  const float* __restrict__ stats,
                                  const float* __restrict__ t1_w,  // [64,5]
                                  const float* __restrict__ t1_b,  // [64]
                                  const float* __restrict__ t2_w,  // [128,64]
                                  const float* __restrict__ t2_b,  // [128]
                                  const float* __restrict__ node_prompted,
                                  const float* __restrict__ edge_agg,
                                  float* __restrict__ final_x,
                                  int N) {
  const float dmax  = __uint_as_float(((const unsigned int*)stats)[0]);
  const float dsum  = stats[1];
  const float dmean = dsum / (float)N;
  int n = blockIdx.x * blockDim.x + threadIdx.x;
  int stride = gridDim.x * blockDim.x;
  for (; n < N; n += stride) {
    float deg = degree[n];
    float f[5];
    f[0] = deg / (dmax + 1e-6f);
    f[1] = hash_uniform((unsigned)n, 0xA511E9B3u) * 0.5f + 0.25f;
    f[2] = deg / (dsum + 1e-6f);
    f[3] = 1.0f / (1.0f + __expf(-(deg - dmean)));
    f[4] = hash_uniform((unsigned)n, 0x63D83595u);

    float h[HID];
#pragma unroll
    for (int j = 0; j < HID; ++j) {
      float acc = t1_b[j];
#pragma unroll
      for (int k = 0; k < 5; ++k) acc = fmaf(t1_w[j * 5 + k], f[k], acc);
      h[j] = fmaxf(acc, 0.0f);
    }
    const float* np = node_prompted + (size_t)n * DIM;
    const float* ea = edge_agg + (size_t)n * DIM;
    float* out = final_x + (size_t)n * DIM;
    for (int c2 = 0; c2 < DIM; ++c2) {
      float acc = t2_b[c2];
#pragma unroll
      for (int j = 0; j < HID; ++j) acc = fmaf(t2_w[c2 * HID + j], h[j], acc);
      out[c2] = np[c2] + acc * ea[c2];
    }
  }
}

// ---------------- launch ----------------
extern "C" void kernel_launch(void* const* d_in, const int* in_sizes, int n_in,
                              void* d_out, int out_size, void* d_ws, size_t ws_size,
                              hipStream_t stream) {
  const float* x             = (const float*)d_in[0];
  const int*   edge_index    = (const int*)d_in[1];
  // d_in[2] = layer (unused)
  const float* anchor_prompt = (const float*)d_in[3];
  const float* w_weight      = (const float*)d_in[4];
  const float* w_bias        = (const float*)d_in[5];
  const float* node_anchor   = (const float*)d_in[6];
  const float* attn_weight   = (const float*)d_in[7];
  const float* attn_bias     = (const float*)d_in[8];
  const float* t1_w          = (const float*)d_in[9];
  const float* t1_b          = (const float*)d_in[10];
  const float* t2_w          = (const float*)d_in[11];
  const float* t2_b          = (const float*)d_in[12];

  const int N = in_sizes[0] / DIM;
  const int E = in_sizes[1] / 2;
  const int* src = edge_index;
  const int* dst = edge_index + E;

  // output: final_x [N,128] then edge_prompt [E,128]
  float* out_final = (float*)d_out;
  float* out_edge_prompt = out_final + (size_t)N * DIM;

  // workspace layout
  float* node_prompted = (float*)d_ws;                      // N*128
  float* edge_agg      = node_prompted + (size_t)N * DIM;   // N*128
  float* degree        = edge_agg + (size_t)N * DIM;        // N
  float* stats         = degree + N;                        // 2 (max-bits, sum)

  size_t nzero = (size_t)N * DIM + (size_t)N + 2;           // edge_agg..stats
  zero_kernel<<<512, 256, 0, stream>>>(edge_agg, nzero);
  degree_kernel<<<1024, 256, 0, stream>>>(src, degree, E);
  stats_kernel<<<256, 256, 0, stream>>>(degree, stats, N);
  node_kernel<<<(N + 63) / 64, 128, 0, stream>>>(x, attn_weight, attn_bias, node_anchor,
                                                 node_prompted, N);
  edge_kernel<<<2500, 128, 0, stream>>>(x, src, dst, w_weight, w_bias, anchor_prompt,
                                        out_edge_prompt, edge_agg, E);
  topo_final_kernel<<<512, 256, 0, stream>>>(degree, stats, t1_w, t1_b, t2_w, t2_b,
                                             node_prompted, edge_agg, out_final, N);
}
